// WordEncoder_7172595384971
// MI455X (gfx1250) — matrix operand order repaired and checked by reading, not compiled, not run on hardware
//
#include <hip/hip_runtime.h>

// ---------------------------------------------------------------------------
// WordEncoder on MI455X (gfx1250, wave32).
// Compute-bound: ~137 GFLOP of GEMM vs ~100MB HBM traffic (~4.3us @23.3TB/s),
// so all matmuls (QKV proj, QK^T, PV, out proj) run on
// v_wmma_f32_16x16x32_bf16 (fp32 accum). Attention is flash-style so the
// 2048x2048 score matrices never leave LDS. Tile fills use CDNA5
// GLOBAL_LOAD_ASYNC_TO_LDS_B128 (ASYNCcnt) instead of VGPR round trips.
// ---------------------------------------------------------------------------

#define BB 4
#define SS 2048
#define DD 1024
#define HH 16
#define DH 64
#define NW 400

typedef __bf16 bf16;
typedef __attribute__((ext_vector_type(16))) __bf16 v16bf;
typedef __attribute__((ext_vector_type(8)))  float  v8f;

// ---- CDNA5 async copy: global -> LDS, 16B per lane, tracked by ASYNCcnt ----
__device__ __forceinline__ unsigned lds_off(const void* p) {
  // generic pointer to __shared__: low 32 bits are the wave-relative LDS
  // address (ISA 10.2: LDS aperture uses addr[31:0])
  return (unsigned)(size_t)p;
}
__device__ __forceinline__ void async_copy16(void* lds_dst, const void* gsrc) {
  asm volatile("global_load_async_to_lds_b128 %0, %1, off"
               :: "v"(lds_off(lds_dst)), "v"(gsrc)
               : "memory");
}
__device__ __forceinline__ void wait_async() {
  asm volatile("s_wait_asynccnt 0x0" ::: "memory");
}

// ---- WMMA fragment loaders (layouts per ISA 7.12.2, wave32) ----------------

// A fragment: 16x32 bf16 (M x K), source row-major [m][k], stride lda.
__device__ __forceinline__ v16bf load_frag_a(const bf16* src, int lda) {
  const int lane = threadIdx.x & 31;
  const int hf   = lane >> 4;
  const int row  = lane & 15;
  v16bf a;
#pragma unroll
  for (int e = 0; e < 16; ++e) {
    const int chunk = ((e >> 3) << 1) | hf;   // 0..3
    const int k     = (e & 7) + (chunk << 3); // 0..31
    a[e] = src[row * lda + k];
  }
  return a;
}

// B fragment from transposed storage [n][k] (stride ldb): B[k][n] = src[n*ldb+k]
// Contiguous in k -> compiler vectorizes to 2x ds_load_b128 per lane.
__device__ __forceinline__ v16bf load_frag_b_t(const bf16* src, int ldb) {
  const int lane = threadIdx.x & 31;
  const int g    = lane >> 4;
  const int n    = lane & 15;
  v16bf b;
#pragma unroll
  for (int e = 0; e < 16; ++e) {
    const int k = (e & 7) + ((((e >> 3) << 1) | g) << 3);
    b[e] = src[n * ldb + k];
  }
  return b;
}

#define WMMA_BF16(A, Bv, C) \
  __builtin_amdgcn_wmma_f32_16x16x32_bf16(false, (A), false, (Bv), (short)0, (C), false, false)

// ---------------------------------------------------------------------------
// K0a: weights fp32 -> bf16
__global__ __launch_bounds__(256) void cvt_kernel(const float* __restrict__ w_in,
                                                  const float* __restrict__ w_out,
                                                  bf16* __restrict__ wi,
                                                  bf16* __restrict__ wo) {
  const int i = blockIdx.x * 256 + threadIdx.x;
  if (i < 3 * DD * DD) wi[i] = (bf16)w_in[i];
  if (i < DD * DD)     wo[i] = (bf16)w_out[i];
}

// K0b: seg[pos] = #boundaries <= pos  (searchsorted right)
__global__ __launch_bounds__(256) void seg_kernel(const int* __restrict__ boundaries,
                                                  int* __restrict__ seg) {
  const int pos = blockIdx.x * 256 + threadIdx.x;
  if (pos >= SS) return;
  int lo = 0, hi = NW;
  while (lo < hi) {
    int mid = (lo + hi) >> 1;
    if (boundaries[mid] <= pos) lo = mid + 1; else hi = mid;
  }
  seg[pos] = lo;
}

// ---------------------------------------------------------------------------
// K1: x = LN(char + pattern_emb[patterns]); write fp32 xln and bf16 xb
__global__ __launch_bounds__(256) void embed_ln_kernel(const float* __restrict__ cx,
                                                       const int* __restrict__ patterns,
                                                       const float* __restrict__ pemb,
                                                       const float* __restrict__ gamma,
                                                       const float* __restrict__ beta,
                                                       float* __restrict__ xln,
                                                       bf16* __restrict__ xb) {
  const int row = blockIdx.x;           // b*S + s
  const int s   = row & (SS - 1);
  const int tid = threadIdx.x;
  const int pat = patterns[s];
  __shared__ float red[256];

  float vals[4];
  float sum = 0.f;
#pragma unroll
  for (int i = 0; i < 4; ++i) {
    const int e = tid + i * 256;
    vals[i] = cx[(size_t)row * DD + e] + pemb[pat * DD + e];
    sum += vals[i];
  }
  red[tid] = sum; __syncthreads();
  for (int st = 128; st > 0; st >>= 1) {
    if (tid < st) red[tid] += red[tid + st];
    __syncthreads();
  }
  const float mu = red[0] * (1.0f / DD);
  __syncthreads();

  float vs = 0.f;
#pragma unroll
  for (int i = 0; i < 4; ++i) { const float d = vals[i] - mu; vs += d * d; }
  red[tid] = vs; __syncthreads();
  for (int st = 128; st > 0; st >>= 1) {
    if (tid < st) red[tid] += red[tid + st];
    __syncthreads();
  }
  const float var  = red[0] * (1.0f / DD);
  const float rstd = rsqrtf(var + 1e-5f);

#pragma unroll
  for (int i = 0; i < 4; ++i) {
    const int e = tid + i * 256;
    const float y = (vals[i] - mu) * rstd * gamma[e] + beta[e];
    xln[(size_t)row * DD + e] = y;
    xb [(size_t)row * DD + e] = (bf16)y;
  }
}

// ---------------------------------------------------------------------------
// K2: QKV GEMM: [8192 x 1024] x [1024 x 3072] -> q/k/v as [B,H,S,DH] bf16.
// 64x64 tile per block, 4 waves; K-step 64 = 8 WMMAs per barrier pair.
__global__ __launch_bounds__(128) void qkv_gemm_kernel(const bf16* __restrict__ xb,
                                                       const bf16* __restrict__ w,
                                                       const float* __restrict__ b_in,
                                                       bf16* __restrict__ qo,
                                                       bf16* __restrict__ ko,
                                                       bf16* __restrict__ vo) {
  __shared__ __align__(16) bf16 At[64 * 72];
  __shared__ __align__(16) bf16 Bt[64 * 72];
  const int tid  = threadIdx.x;
  const int wv   = tid >> 5;
  const int lane = tid & 31;
  const int hf   = lane >> 4;
  const int col  = lane & 15;
  const int m0   = blockIdx.x * 64;
  const int n0   = blockIdx.y * 64;

  v8f acc[4];
#pragma unroll
  for (int j = 0; j < 4; ++j)
#pragma unroll
    for (int e = 0; e < 8; ++e) acc[j][e] = 0.f;

  for (int k0 = 0; k0 < DD; k0 += 64) {
    __syncthreads();   // LDS reuse guard (dscnt flushed by barrier lowering)
#pragma unroll
    for (int i = 0; i < 4; ++i) {
      const int chunk = tid + i * 128;     // 0..511: 64 rows x 8 16B-chunks
      const int r = chunk >> 3;
      const int c = (chunk & 7) << 3;
      async_copy16(&At[r * 72 + c], &xb[(size_t)(m0 + r) * DD + k0 + c]);
      async_copy16(&Bt[r * 72 + c], &w [(size_t)(n0 + r) * DD + k0 + c]);
    }
    wait_async();
    __syncthreads();

    const bf16* abase = &At[(wv * 16) * 72];
    const v16bf a0 = load_frag_a(abase,      72);
    const v16bf a1 = load_frag_a(abase + 32, 72);
#pragma unroll
    for (int j = 0; j < 4; ++j) {
      const v16bf b0 = load_frag_b_t(&Bt[(j * 16) * 72],      72);
      const v16bf b1 = load_frag_b_t(&Bt[(j * 16) * 72 + 32], 72);
      acc[j] = WMMA_BF16(a0, b0, acc[j]);
      acc[j] = WMMA_BF16(a1, b1, acc[j]);
    }
  }

  // epilogue: + bias, scatter to q/k/v [B,H,S,DH]
#pragma unroll
  for (int j = 0; j < 4; ++j) {
#pragma unroll
    for (int r = 0; r < 8; ++r) {
      const int m = m0 + wv * 16 + r + 8 * hf;
      const int n = n0 + j * 16 + col;
      const float v = acc[j][r] + b_in[n];
      const int t  = n >> 10;         // 0=q,1=k,2=v
      const int d  = n & 1023;
      const int h  = d >> 6;
      const int dh = d & 63;
      const int b  = m >> 11;         // / S
      const int s  = m & (SS - 1);
      bf16* dst = (t == 0) ? qo : ((t == 1) ? ko : vo);
      dst[(((size_t)b * HH + h) * SS + s) * DH + dh] = (bf16)v;
    }
  }
}

// ---------------------------------------------------------------------------
// K3: flash attention per (b,h). Block = 128 q rows, 8 waves (16 rows each),
// 32-key tiles, online softmax, word-mask additive bias (+1.0 same-segment).
__global__ __launch_bounds__(256) void attn_kernel(const bf16* __restrict__ q,
                                                   const bf16* __restrict__ k,
                                                   const bf16* __restrict__ v,
                                                   const int* __restrict__ seg,
                                                   bf16* __restrict__ attn_o) {
  __shared__ __align__(16) bf16 Qs[128 * 72];
  __shared__ __align__(16) bf16 Ks[32 * 72];
  __shared__ __align__(16) bf16 Vt[64 * 40];   // V transposed: [dh][key]
  __shared__ __align__(16) bf16 Ps[8][16 * 36];
  __shared__ int segq[128];
  __shared__ int segk[32];

  const int tid  = threadIdx.x;
  const int wv   = tid >> 5;
  const int lane = tid & 31;
  const int hf   = lane >> 4;
  const int col  = lane & 15;
  const int bh   = blockIdx.x;
  const int b    = bh >> 4;
  const int h    = bh & (HH - 1);
  const int q0   = blockIdx.y * 128;
  const size_t headoff = (size_t)(b * HH + h) * SS * DH;

  // async-load Q tile 128x64 (1024 16B chunks, 4 per thread)
#pragma unroll
  for (int i = 0; i < 4; ++i) {
    const int chunk = tid + i * 256;
    const int r = chunk >> 3;
    const int c = (chunk & 7) << 3;
    async_copy16(&Qs[r * 72 + c], &q[headoff + (size_t)(q0 + r) * DH + c]);
  }
  if (tid < 128) segq[tid] = seg[q0 + tid];
  wait_async();
  __syncthreads();

  float rowmax[8], rowsum[8];
  v8f o[4];
#pragma unroll
  for (int r = 0; r < 8; ++r) { rowmax[r] = -3.0e38f; rowsum[r] = 0.f; }
#pragma unroll
  for (int j = 0; j < 4; ++j)
#pragma unroll
    for (int e = 0; e < 8; ++e) o[j][e] = 0.f;

  const float scale = 0.125f;  // 1/sqrt(64)

  for (int kb = 0; kb < SS; kb += 32) {
    __syncthreads();           // guard Ks/Vt reuse
    {
      const int r = tid >> 3;            // key row 0..31
      const int c = (tid & 7) << 3;      // dh base
      async_copy16(&Ks[r * 72 + c], &k[headoff + (size_t)(kb + r) * DH + c]);
      // V: load 8 dh of one key and transpose-store into Vt[dh][key]
      const uint4 vv = *(const uint4*)&v[headoff + (size_t)(kb + r) * DH + c];
      const bf16* pv = (const bf16*)&vv;
#pragma unroll
      for (int e = 0; e < 8; ++e) Vt[(c + e) * 40 + r] = pv[e];
    }
    if (tid < 32) segk[tid] = seg[kb + tid];
    wait_async();
    __syncthreads();

    // ---- scores: S = Q K^T * scale + mask ----
    const v16bf a0 = load_frag_a(&Qs[(wv * 16) * 72],      72);
    const v16bf a1 = load_frag_a(&Qs[(wv * 16) * 72 + 32], 72);
    float f[2][8];
#pragma unroll
    for (int n = 0; n < 2; ++n) {
      const v16bf b0 = load_frag_b_t(&Ks[(n * 16) * 72],      72);
      const v16bf b1 = load_frag_b_t(&Ks[(n * 16) * 72 + 32], 72);
      v8f sc;
#pragma unroll
      for (int e = 0; e < 8; ++e) sc[e] = 0.f;
      sc = WMMA_BF16(a0, b0, sc);
      sc = WMMA_BF16(a1, b1, sc);
      const int sk = segk[n * 16 + col];
#pragma unroll
      for (int r = 0; r < 8; ++r) {
        const int sq = segq[wv * 16 + r + 8 * hf];
        const float bias = (sq > 0 && sq == sk) ? 1.0f : 0.0f;
        f[n][r] = sc[r] * scale + bias;
      }
    }

    // ---- online softmax (row stats live across each 16-lane half) ----
    float t[8];
#pragma unroll
    for (int r = 0; r < 8; ++r) t[r] = fmaxf(f[0][r], f[1][r]);
#pragma unroll
    for (int d = 1; d < 16; d <<= 1)
#pragma unroll
      for (int r = 0; r < 8; ++r) t[r] = fmaxf(t[r], __shfl_xor(t[r], d, 32));

    float alpha[8], ps[8];
#pragma unroll
    for (int r = 0; r < 8; ++r) {
      const float mnew = fmaxf(rowmax[r], t[r]);
      alpha[r] = __expf(rowmax[r] - mnew);
      f[0][r]  = __expf(f[0][r] - mnew);
      f[1][r]  = __expf(f[1][r] - mnew);
      ps[r]    = f[0][r] + f[1][r];
      rowmax[r] = mnew;
    }
#pragma unroll
    for (int d = 1; d < 16; d <<= 1)
#pragma unroll
      for (int r = 0; r < 8; ++r) ps[r] += __shfl_xor(ps[r], d, 32);
#pragma unroll
    for (int r = 0; r < 8; ++r) rowsum[r] = rowsum[r] * alpha[r] + ps[r];

#pragma unroll
    for (int j = 0; j < 4; ++j)
#pragma unroll
      for (int r = 0; r < 8; ++r) o[j][r] *= alpha[r];

    // stage P (C-frag layout -> row-major LDS) for use as A fragment
#pragma unroll
    for (int n = 0; n < 2; ++n)
#pragma unroll
      for (int r = 0; r < 8; ++r)
        Ps[wv][(r + 8 * hf) * 36 + n * 16 + col] = (bf16)f[n][r];
    __syncthreads();

    // ---- O += P V  (B frags from transposed Vt: contiguous ds_load_b128) ----
    const v16bf ap = load_frag_a(&Ps[wv][0], 36);
#pragma unroll
    for (int j = 0; j < 4; ++j) {
      const v16bf bv = load_frag_b_t(&Vt[(j * 16) * 40], 40);
      o[j] = WMMA_BF16(ap, bv, o[j]);
    }
  }

  // ---- normalize and store [B,S,D] bf16 ----
#pragma unroll
  for (int j = 0; j < 4; ++j) {
#pragma unroll
    for (int r = 0; r < 8; ++r) {
      const float val = o[j][r] / rowsum[r];
      const int srow = q0 + wv * 16 + r + 8 * hf;
      const int d    = h * 64 + j * 16 + col;
      attn_o[((size_t)b * SS + srow) * DD + d] = (bf16)val;
    }
  }
}

// ---------------------------------------------------------------------------
// K4: out proj + bias + residual: out = attn_o @ w_out^T + b_out + xln (fp32)
__global__ __launch_bounds__(128) void out_gemm_kernel(const bf16* __restrict__ ao,
                                                       const bf16* __restrict__ w,
                                                       const float* __restrict__ b_out,
                                                       const float* __restrict__ xln,
                                                       float* __restrict__ out) {
  __shared__ __align__(16) bf16 At[64 * 72];
  __shared__ __align__(16) bf16 Bt[64 * 72];
  const int tid  = threadIdx.x;
  const int wv   = tid >> 5;
  const int lane = tid & 31;
  const int hf   = lane >> 4;
  const int col  = lane & 15;
  const int m0   = blockIdx.x * 64;
  const int n0   = blockIdx.y * 64;

  v8f acc[4];
#pragma unroll
  for (int j = 0; j < 4; ++j)
#pragma unroll
    for (int e = 0; e < 8; ++e) acc[j][e] = 0.f;

  for (int k0 = 0; k0 < DD; k0 += 64) {
    __syncthreads();
#pragma unroll
    for (int i = 0; i < 4; ++i) {
      const int chunk = tid + i * 128;
      const int r = chunk >> 3;
      const int c = (chunk & 7) << 3;
      async_copy16(&At[r * 72 + c], &ao[(size_t)(m0 + r) * DD + k0 + c]);
      async_copy16(&Bt[r * 72 + c], &w [(size_t)(n0 + r) * DD + k0 + c]);
    }
    wait_async();
    __syncthreads();

    const bf16* abase = &At[(wv * 16) * 72];
    const v16bf a0 = load_frag_a(abase,      72);
    const v16bf a1 = load_frag_a(abase + 32, 72);
#pragma unroll
    for (int j = 0; j < 4; ++j) {
      const v16bf b0 = load_frag_b_t(&Bt[(j * 16) * 72],      72);
      const v16bf b1 = load_frag_b_t(&Bt[(j * 16) * 72 + 32], 72);
      acc[j] = WMMA_BF16(a0, b0, acc[j]);
      acc[j] = WMMA_BF16(a1, b1, acc[j]);
    }
  }

#pragma unroll
  for (int j = 0; j < 4; ++j) {
#pragma unroll
    for (int r = 0; r < 8; ++r) {
      const int m = m0 + wv * 16 + r + 8 * hf;
      const int n = n0 + j * 16 + col;
      out[(size_t)m * DD + n] = acc[j][r] + b_out[n] + xln[(size_t)m * DD + n];
    }
  }
}

// ---------------------------------------------------------------------------
extern "C" void kernel_launch(void* const* d_in, const int* in_sizes, int n_in,
                              void* d_out, int out_size, void* d_ws, size_t ws_size,
                              hipStream_t stream) {
  const float* char_repr  = (const float*)d_in[0];
  const int*   patterns   = (const int*)  d_in[1];
  const int*   boundaries = (const int*)  d_in[2];
  const float* pattern_emb= (const float*)d_in[3];
  const float* gamma      = (const float*)d_in[4];
  const float* beta       = (const float*)d_in[5];
  const float* w_in       = (const float*)d_in[6];
  const float* b_in       = (const float*)d_in[7];
  const float* w_out      = (const float*)d_in[8];
  const float* b_out      = (const float*)d_in[9];
  float* out = (float*)d_out;

  // workspace carve-out
  char* ws = (char*)d_ws;
  size_t off = 0;
  auto carve = [&](size_t bytes) -> char* {
    char* p = ws + off;
    off = (off + bytes + 255) & ~(size_t)255;
    return p;
  };
  const size_t BSD = (size_t)BB * SS * DD;
  float* xln    = (float*)carve(BSD * 4);
  bf16*  xb     = (bf16*) carve(BSD * 2);
  bf16*  wi_b   = (bf16*) carve((size_t)3 * DD * DD * 2);
  bf16*  wo_b   = (bf16*) carve((size_t)DD * DD * 2);
  bf16*  qbuf   = (bf16*) carve(BSD * 2);
  bf16*  kbuf   = (bf16*) carve(BSD * 2);
  bf16*  vbuf   = (bf16*) carve(BSD * 2);
  bf16*  aobuf  = (bf16*) carve(BSD * 2);
  int*   segbuf = (int*)  carve(SS * 4);
  (void)ws_size; (void)in_sizes; (void)n_in; (void)out_size;

  cvt_kernel<<<dim3((3 * DD * DD + 255) / 256), dim3(256), 0, stream>>>(w_in, w_out, wi_b, wo_b);
  seg_kernel<<<dim3((SS + 255) / 256), dim3(256), 0, stream>>>(boundaries, segbuf);
  embed_ln_kernel<<<dim3(BB * SS), dim3(256), 0, stream>>>(char_repr, patterns, pattern_emb,
                                                           gamma, beta, xln, xb);
  qkv_gemm_kernel<<<dim3((BB * SS) / 64, (3 * DD) / 64), dim3(128), 0, stream>>>(
      xb, wi_b, b_in, qbuf, kbuf, vbuf);
  attn_kernel<<<dim3(BB * HH, SS / 128), dim3(256), 0, stream>>>(qbuf, kbuf, vbuf, segbuf, aobuf);
  out_gemm_kernel<<<dim3((BB * SS) / 64, DD / 64), dim3(128), 0, stream>>>(
      aobuf, wo_b, b_out, xln, out);
}
